// ELFEB_69234872812261
// MI455X (gfx1250) — matrix operand order repaired and checked by reading, not compiled, not run on hardware
//
#include <hip/hip_runtime.h>
#include <hip/hip_bf16.h>
#include <math.h>

// Problem constants (match reference)
#define CCH 256
#define BB  8
#define HH  128
#define WWI 128
#define HWH (HH * WWI)        // 16384 pixels per image
#define GSH (CCH / 5)         // 51: shift group size

// GEMM tiling
#define KTILE 32              // K per WMMA chain step (bf16 16x16x32)
#define NTILE 64              // pixels per workgroup tile
#define KPAD  40              // LDS row pitch in halves (80B: bank-spread, 16B aligned)
#define NKT   (CCH / KTILE)   // 8 K-steps

typedef __attribute__((ext_vector_type(16))) __bf16 v16bf;
typedef __attribute__((ext_vector_type(8)))  __bf16 v8bf;
typedef __attribute__((ext_vector_type(8)))  float  v8f;

// Per-input-channel shift from the frozen one-hot 3x3 depthwise weights.
__device__ __forceinline__ void shift_of(int c, int& dy, int& dx) {
    if (c < GSH)            { dy = 0;  dx = 1;  }   // from right
    else if (c < 2 * GSH)   { dy = 0;  dx = -1; }   // from left
    else if (c < 3 * GSH)   { dy = 1;  dx = 0;  }   // from below
    else if (c < 4 * GSH)   { dy = -1; dx = 0;  }   // from above
    else                    { dy = 0;  dx = 0;  }   // identity
}

// ---------------------------------------------------------------------------
// Kernel 0: split the two 256x256 1x1-conv weight matrices into bf16 hi/lo.
// hi = rne(bf16(w)); lo = rne(bf16(w - f32(hi)))  =>  hi+lo ~ w to ~16 mantissa bits.
// ---------------------------------------------------------------------------
__global__ void wprep_kernel(const float* __restrict__ w1, const float* __restrict__ w2,
                             __bf16* __restrict__ w1h, __bf16* __restrict__ w1l,
                             __bf16* __restrict__ w2h, __bf16* __restrict__ w2l) {
    int i = blockIdx.x * 256 + threadIdx.x;          // 0 .. 65535
    float a = w1[i];
    __bf16 ah = (__bf16)a;
    __bf16 al = (__bf16)(a - (float)ah);
    w1h[i] = ah; w1l[i] = al;
    float b = w2[i];
    __bf16 bh = (__bf16)b;
    __bf16 bl = (__bf16)(b - (float)bh);
    w2h[i] = bh; w2l[i] = bl;
}

// ---- staging helpers (shift folded into global read; zero-pad at borders) --
__device__ __forceinline__ void load_slab(const float* __restrict__ srcB, int cBase,
                                          int cLocal, int h0, int w0, int pxOct,
                                          float v[8]) {
    const int c = cBase + cLocal;
    int dy, dx; shift_of(c, dy, dx);
    const int hs = h0 + dy;
    const bool rowOK = (hs >= 0) && (hs < HH);
    const float* rowp = srcB + (size_t)c * HWH + hs * WWI;
    #pragma unroll
    for (int i = 0; i < 8; ++i) {
        const int ws = w0 + pxOct + i + dx;
        v[i] = (rowOK && ws >= 0 && ws < WWI) ? rowp[ws] : 0.0f;
    }
}

__device__ __forceinline__ void store_slab(__bf16 (*__restrict__ XH)[KPAD],
                                           __bf16 (*__restrict__ XL)[KPAD],
                                           int cLocal, int pxOct, const float v[8]) {
    #pragma unroll
    for (int i = 0; i < 8; ++i) {
        __bf16 hv = (__bf16)v[i];
        XH[pxOct + i][cLocal] = hv;
        XL[pxOct + i][cLocal] = (__bf16)(v[i] - (float)hv);
    }
}

// ---------------------------------------------------------------------------
// Kernel G: out[b,o,p] = sum_c W[o,c] * src[b,c, shift_c(p)] + bias[o] (+ x[b,o,p])
// Workgroup: 256 threads (8 waves). Tile: M=256 (all out channels) x N=64 pixels.
// Wave w owns M rows [32w,32w+32): 2 (M) x 4 (N) wmma tiles, K chained 8x32.
// Split-bf16: acc = wmma(Ah,Bh) ; wmma(Ah,Bl) ; wmma(Al,Bh)  (same f32 acc).
// Double-buffered LDS slab, one barrier per K-step. A-fragment loads are issued
// BEFORE the next-slab staging loads so s_wait_loadcnt for the WMMA operands
// leaves the (in-order-counted) HBM staging loads still in flight.
// ---------------------------------------------------------------------------
template <bool ADD_SHORTCUT>
__global__ void __launch_bounds__(256)
gemm_shift_kernel(const float*  __restrict__ src,      // [B][C][H][W], shift applied on read
                  const __bf16* __restrict__ Wh,
                  const __bf16* __restrict__ Wl,
                  const float*  __restrict__ bias,
                  const float*  __restrict__ shortcut, // x (only if ADD_SHORTCUT)
                  float*        __restrict__ dst) {
    __shared__ __bf16 XsH[2][NTILE][KPAD];
    __shared__ __bf16 XsL[2][NTILE][KPAD];

    const int tid  = threadIdx.x;
    const int lane = tid & 31;
    const int wave = tid >> 5;                 // 0..7
    const int blk  = blockIdx.x;               // 0..2047
    const int b    = blk >> 8;                 // image
    const int nt   = blk & 255;                // 64-pixel tile (one half-row, 64|128)
    const int pixBase = nt * NTILE;
    const int h0 = pixBase >> 7;
    const int w0 = pixBase & (WWI - 1);

    const float* srcB = src + (size_t)b * CCH * HWH;

    // cooperative-loader mapping: 32 channels x 64 pixels, 8 elems/thread
    const int cLocal = tid >> 3;               // 0..31
    const int pxOct  = (tid & 7) * 8;          // 0,8,..,56

    const int mBase = wave * 32;

    v8f acc[2][4];
    #pragma unroll
    for (int mi = 0; mi < 2; ++mi)
        #pragma unroll
        for (int ni = 0; ni < 4; ++ni)
            #pragma unroll
            for (int j = 0; j < 8; ++j) acc[mi][ni][j] = 0.0f;

    const int mrow  = lane & 15;
    const int kaoff = (lane < 16) ? 0 : 8;     // A frag: 16-bit A 16x32 layout
    const int kboff = (lane < 16) ? 0 : 8;     // B frag: same K layout as A frag

    // prologue: stage slab 0
    {
        float cur[8];
        load_slab(srcB, 0, cLocal, h0, w0, pxOct, cur);
        store_slab(XsH[0], XsL[0], cLocal, pxOct, cur);
    }
    __syncthreads();

    for (int kt = 0; kt < NKT; ++kt) {
        const int cBase = kt * KTILE;
        const int buf = kt & 1;

        // ---- A fragments first (L2-hot weights; waits won't drain staging) ----
        v16bf aH[2], aL[2];
        #pragma unroll
        for (int mi = 0; mi < 2; ++mi) {
            const int m = mBase + mi * 16 + mrow;
            const __bf16* ph = Wh + (size_t)m * CCH + cBase;
            const __bf16* pl = Wl + (size_t)m * CCH + cBase;
            v8bf h0v = *(const v8bf*)(ph + kaoff);
            v8bf h1v = *(const v8bf*)(ph + 16 + kaoff);
            v8bf l0v = *(const v8bf*)(pl + kaoff);
            v8bf l1v = *(const v8bf*)(pl + 16 + kaoff);
            #pragma unroll
            for (int j = 0; j < 8; ++j) {
                aH[mi][j] = h0v[j]; aH[mi][8 + j] = h1v[j];
                aL[mi][j] = l0v[j]; aL[mi][8 + j] = l1v[j];
            }
        }

        // ---- then issue next-slab HBM loads (latency hidden under WMMAs) ----
        float nxt[8];
        if (kt + 1 < NKT)
            load_slab(srcB, cBase + KTILE, cLocal, h0, w0, pxOct, nxt);

        // ---- B fragments from LDS + WMMA chain ----
        #pragma unroll
        for (int ni = 0; ni < 4; ++ni) {
            const int n = ni * 16 + mrow;
            v8bf bh0 = *(const v8bf*)(&XsH[buf][n][kboff]);
            v8bf bh1 = *(const v8bf*)(&XsH[buf][n][kboff + 16]);
            v8bf bl0 = *(const v8bf*)(&XsL[buf][n][kboff]);
            v8bf bl1 = *(const v8bf*)(&XsL[buf][n][kboff + 16]);
            v16bf bH, bL;
            #pragma unroll
            for (int j = 0; j < 8; ++j) {
                bH[j] = bh0[j]; bH[8 + j] = bh1[j];
                bL[j] = bl0[j]; bL[8 + j] = bl1[j];
            }
            #pragma unroll
            for (int mi = 0; mi < 2; ++mi) {
                acc[mi][ni] = __builtin_amdgcn_wmma_f32_16x16x32_bf16(
                    false, aH[mi], false, bH, (short)0, acc[mi][ni], false, false);
                acc[mi][ni] = __builtin_amdgcn_wmma_f32_16x16x32_bf16(
                    false, aH[mi], false, bL, (short)0, acc[mi][ni], false, false);
                acc[mi][ni] = __builtin_amdgcn_wmma_f32_16x16x32_bf16(
                    false, aL[mi], false, bH, (short)0, acc[mi][ni], false, false);
            }
        }

        // convert + store next slab into the alternate buffer, then fence.
        // (the single barrier both publishes these stores and closes out all
        //  reads of buf^1 from the previous iteration)
        if (kt + 1 < NKT)
            store_slab(XsH[buf ^ 1], XsL[buf ^ 1], cLocal, pxOct, nxt);
        __syncthreads();
    }

    // ---- epilogue: C/D layout -> lanes 0-15: M=j, lanes 16-31: M=8+j; N=lane&15
    const int mOff = (lane >> 4) * 8;
    float* dstB = dst + (size_t)b * CCH * HWH;
    const float* shB = ADD_SHORTCUT ? (shortcut + (size_t)b * CCH * HWH) : nullptr;
    #pragma unroll
    for (int mi = 0; mi < 2; ++mi) {
        #pragma unroll
        for (int ni = 0; ni < 4; ++ni) {
            const int p = pixBase + ni * 16 + mrow;
            #pragma unroll
            for (int j = 0; j < 8; ++j) {
                const int o = mBase + mi * 16 + mOff + j;
                float v = acc[mi][ni][j] + bias[o];
                if (ADD_SHORTCUT) v += shB[(size_t)o * HWH + p];
                dstB[(size_t)o * HWH + p] = v;
            }
        }
    }
}

// ---------------------------------------------------------------------------
// Kernel D: depthwise 5x5 + bias, then exact GELU. Memory bound.
// ---------------------------------------------------------------------------
__global__ void __launch_bounds__(256)
dw5_gelu_kernel(const float* __restrict__ y1, const float* __restrict__ dww,
                const float* __restrict__ dwb, float* __restrict__ y2) {
    const int idx = blockIdx.x * 256 + threadIdx.x;
    const int p = idx & (HWH - 1);
    const int c = (idx >> 14) & (CCH - 1);
    const int b = idx >> 22;
    const int h = p >> 7, w = p & (WWI - 1);
    const float* in = y1 + ((size_t)b * CCH + c) * HWH;
    const float* wt = dww + c * 25;
    float s = dwb[c];
    #pragma unroll
    for (int i = 0; i < 5; ++i) {
        const int hh = h + i - 2;
        if (hh < 0 || hh >= HH) continue;
        #pragma unroll
        for (int j = 0; j < 5; ++j) {
            const int wc = w + j - 2;
            if (wc < 0 || wc >= WWI) continue;
            s += in[hh * WWI + wc] * wt[i * 5 + j];
        }
    }
    // exact GELU: 0.5*x*(1+erf(x/sqrt(2)))
    y2[idx] = 0.5f * s * (1.0f + erff(s * 0.70710678118654752f));
}

// ---------------------------------------------------------------------------
// Launch: wprep -> GEMM1(shift(x)) -> dw5x5+GELU -> GEMM2(shift(y2)) + residual
// ---------------------------------------------------------------------------
extern "C" void kernel_launch(void* const* d_in, const int* in_sizes, int n_in,
                              void* d_out, int out_size, void* d_ws, size_t ws_size,
                              hipStream_t stream) {
    (void)in_sizes; (void)n_in; (void)out_size; (void)ws_size;
    const float* x       = (const float*)d_in[0];
    // d_in[1] = w_shift: frozen one-hot shift, realized as index offsets
    const float* conv1_w = (const float*)d_in[2];
    const float* conv1_b = (const float*)d_in[3];
    const float* dw_w    = (const float*)d_in[4];
    const float* dw_b    = (const float*)d_in[5];
    const float* conv2_w = (const float*)d_in[6];
    const float* conv2_b = (const float*)d_in[7];

    const size_t tensorElems = (size_t)BB * CCH * HWH;   // 33,554,432 f32
    float*  y1  = (float*)d_ws;
    float*  y2  = y1 + tensorElems;
    __bf16* w1h = (__bf16*)(y2 + tensorElems);
    __bf16* w1l = w1h + CCH * CCH;
    __bf16* w2h = w1l + CCH * CCH;
    __bf16* w2l = w2h + CCH * CCH;

    // 0) split weights into bf16 hi/lo (64K elems each matrix)
    wprep_kernel<<<(CCH * CCH) / 256, 256, 0, stream>>>(conv1_w, conv2_w,
                                                        w1h, w1l, w2h, w2l);

    // 1) y1 = conv1(shift(x)) + b1    [WMMA GEMM, shift folded into loads]
    const int gemmBlocks = BB * (HWH / NTILE);           // 8 * 256 = 2048
    gemm_shift_kernel<false><<<gemmBlocks, 256, 0, stream>>>(
        x, w1h, w1l, conv1_b, nullptr, y1);

    // 2) y2 = gelu(dw5x5(y1) + b_dw)
    dw5_gelu_kernel<<<(int)(tensorElems / 256), 256, 0, stream>>>(y1, dw_w, dw_b, y2);

    // 3) out = conv2(shift(y2)) + b2 + x
    gemm_shift_kernel<true><<<gemmBlocks, 256, 0, stream>>>(
        y2, w2h, w2l, conv2_b, x, (float*)d_out);
}
